// MHSA_11965778887138
// MI455X (gfx1250) — compile-verified, matrix operand order checked by reading, not run on hardware
//
#include <hip/hip_runtime.h>
#include <hip/hip_bf16.h>

// Problem constants (B=4, C=2048, N=8192, H=8, d=256)
#define BB 4
#define CC 2048
#define NN 8192
#define HH 8
#define DD 256
#define BH (BB * HH)          // 32

typedef __attribute__((ext_vector_type(16))) __bf16 v16bf;
typedef __attribute__((ext_vector_type(8)))  __bf16 v8bf;
typedef __attribute__((ext_vector_type(8)))  float  v8f;
typedef __attribute__((ext_vector_type(4)))  int    v4i;

typedef __attribute__((address_space(1))) v4i* gptr128;   // global 16B chunk
typedef __attribute__((address_space(3))) v4i* lptr128;   // LDS 16B chunk

// Native f32 -> bf16 convert (gfx1250 has hardware cvt; avoid manual bit ops)
__device__ __forceinline__ __bf16 f2bf(float x) { return (__bf16)x; }

// ---------------------------------------------------------------------------
// Kernel 1: per-(b,c) row stats of k over the sequence dim (N) for softmax
// ---------------------------------------------------------------------------
__global__ void __launch_bounds__(256) k_rowstats(const float* __restrict__ k,
                                                  float* __restrict__ kmax,
                                                  float* __restrict__ ksum) {
    __shared__ float red[256];
    const int row = blockIdx.x;                 // 0 .. B*C-1
    const float* rp = k + (size_t)row * NN;
    const int t = threadIdx.x;

    float m = -3.402823e38f;
    for (int n = t; n < NN; n += 256) m = fmaxf(m, rp[n]);
    red[t] = m; __syncthreads();
    for (int s = 128; s > 0; s >>= 1) {
        if (t < s) red[t] = fmaxf(red[t], red[t + s]);
        __syncthreads();
    }
    const float mx = red[0]; __syncthreads();

    float sum = 0.f;
    for (int n = t; n < NN; n += 256) sum += __expf(rp[n] - mx);
    red[t] = sum; __syncthreads();
    for (int s = 128; s > 0; s >>= 1) {
        if (t < s) red[t] += red[t + s];
        __syncthreads();
    }
    if (t == 0) { kmax[row] = mx; ksum[row] = red[0]; }
}

// ---------------------------------------------------------------------------
// Kernel 2: per-(b,h,n) stats of q over the head-feature dim (d, stride N)
// ---------------------------------------------------------------------------
__global__ void __launch_bounds__(256) q_colstats(const float* __restrict__ q,
                                                  float* __restrict__ qmax,
                                                  float* __restrict__ qsum) {
    int blk  = blockIdx.x;                      // B*H * (N/256) = 1024
    const int nblk = blk & 31;  blk >>= 5;
    const int bh   = blk;                       // 0..31
    const int n    = nblk * 256 + threadIdx.x;
    const int cb   = (bh >> 3) * CC + (bh & 7) * DD;
    const float* qp = q + (size_t)cb * NN + n;

    float m = -3.402823e38f;
    for (int i = 0; i < DD; ++i) m = fmaxf(m, qp[(size_t)i * NN]);
    float s = 0.f;
    for (int i = 0; i < DD; ++i) s += __expf(qp[(size_t)i * NN] - m);
    qmax[(size_t)bh * NN + n] = m;
    qsum[(size_t)bh * NN + n] = s;
}

// ---------------------------------------------------------------------------
// Kernel 3: kv[b,h] = softmax(k)^T @ v   (256x256, K = 8192, K-split atomics)
// grid = bh(32) * subM(2) * subN(2) * ksplit(8); block = 256 (8 wave32)
// ---------------------------------------------------------------------------
__global__ void __launch_bounds__(256) gemm1_kv(const float* __restrict__ k,
                                                const float* __restrict__ v,
                                                const float* __restrict__ kmax,
                                                const float* __restrict__ ksum,
                                                float* __restrict__ kv) {
    int blk = blockIdx.x;
    const int ksp = blk & 7;  blk >>= 3;
    const int sn  = blk & 1;  blk >>= 1;
    const int sm  = blk & 1;  blk >>= 1;
    const int bh  = blk;
    const int cb  = (bh >> 3) * CC + (bh & 7) * DD;

    const int wave = threadIdx.x >> 5, lane = threadIdx.x & 31;
    const int wm = wave & 3, wn = wave >> 2;
    const int iW = sm * 128 + wm * 32;          // M (feature i) origin of wave
    const int eW = sn * 128 + wn * 64;          // N (feature e) origin of wave
    const int lr = lane & 15, hp = lane >> 4;

    const float* kb = k + (size_t)cb * NN;
    const float* vb = v + (size_t)cb * NN;

    float mxA[2], rsA[2];
#pragma unroll
    for (int mt = 0; mt < 2; ++mt) {
        const int ir = iW + 16 * mt + lr;
        mxA[mt] = kmax[cb + ir];
        rsA[mt] = 1.0f / ksum[cb + ir];
    }

    v8f acc[2][4] = {};
    const int kk0base = ksp * (NN / 8);

    for (int kk = 0; kk < NN / 8; kk += 32) {
        const int kk0 = kk0base + kk;
        // A fragments: softmaxed k rows, K contiguous (two 32B chunks/lane)
        v16bf aF[2];
#pragma unroll
        for (int mt = 0; mt < 2; ++mt) {
            const float* rp = kb + (size_t)(iW + 16 * mt + lr) * NN + kk0 + 8 * hp;
            __builtin_prefetch(rp + 64, 0, 0);
            float4 c0 = *(const float4*)(rp);
            float4 c1 = *(const float4*)(rp + 4);
            float4 c2 = *(const float4*)(rp + 16);
            float4 c3 = *(const float4*)(rp + 20);
            float f[16] = {c0.x, c0.y, c0.z, c0.w, c1.x, c1.y, c1.z, c1.w,
                           c2.x, c2.y, c2.z, c2.w, c3.x, c3.y, c3.z, c3.w};
#pragma unroll
            for (int j = 0; j < 16; ++j)
                aF[mt][j] = f2bf(__expf(f[j] - mxA[mt]) * rsA[mt]);
        }
        // B fragments: v columns (channel e rows in memory), K contiguous
        v16bf bF[4];
#pragma unroll
        for (int nt = 0; nt < 4; ++nt) {
            const float* rp = vb + (size_t)(eW + 16 * nt + lr) * NN + kk0 + 8 * hp;
            __builtin_prefetch(rp + 64, 0, 0);
            float4 c0 = *(const float4*)(rp);
            float4 c1 = *(const float4*)(rp + 4);
            float4 c2 = *(const float4*)(rp + 16);
            float4 c3 = *(const float4*)(rp + 20);
            float f[16] = {c0.x, c0.y, c0.z, c0.w, c1.x, c1.y, c1.z, c1.w,
                           c2.x, c2.y, c2.z, c2.w, c3.x, c3.y, c3.z, c3.w};
#pragma unroll
            for (int j = 0; j < 16; ++j) bF[nt][j] = f2bf(f[j]);
        }
#pragma unroll
        for (int mt = 0; mt < 2; ++mt)
#pragma unroll
            for (int nt = 0; nt < 4; ++nt)
                acc[mt][nt] = __builtin_amdgcn_wmma_f32_16x16x32_bf16(
                    false, aF[mt], false, bF[nt], (short)0, acc[mt][nt], false, false);
    }

    // Accumulate K-split partials
#pragma unroll
    for (int mt = 0; mt < 2; ++mt)
#pragma unroll
        for (int nt = 0; nt < 4; ++nt) {
            const int e = eW + 16 * nt + lr;
#pragma unroll
            for (int r = 0; r < 8; ++r) {
                const int ir = iW + 16 * mt + r + 8 * hp;   // D: M = r + 8*(lane/16)
                atomicAdd(&kv[((size_t)bh * DD + ir) * DD + e], acc[mt][nt][r]);
            }
        }
}

// ---------------------------------------------------------------------------
// Kernel 4: kvt[bh][e][i] = bf16(kv[bh][i][e])
// ---------------------------------------------------------------------------
__global__ void __launch_bounds__(256) kv_transpose(const float* __restrict__ kv,
                                                    unsigned short* __restrict__ kvt_u16) {
    __bf16* kvt = (__bf16*)kvt_u16;
    const unsigned idx = blockIdx.x * 256u + threadIdx.x;   // < 32*256*256
    const unsigned bh = idx >> 16, rem = idx & 65535u;
    const unsigned e = rem >> 8, i = rem & 255u;
    kvt[idx] = f2bf(kv[((size_t)bh << 16) + ((size_t)i << 8) + e]);
}

// ---------------------------------------------------------------------------
// Kernel 5: out[b, h*d+e, n] = sum_i softmax_i(q)[n,i] * kv[i,e]
// M = n (8192), N = e (256), K = i (256)
// grid = bh(32) * mb(64) * nb(2); block = 256 (8 wave32)
// B operand (kv^T tile, 64 KB) is staged into LDS once per block via the
// CDNA5 async global->LDS path, then served to WMMA with ds_load_b128.
// ---------------------------------------------------------------------------
__global__ void __launch_bounds__(256) gemm2_out(const float* __restrict__ q,
                                                 const float* __restrict__ qmax,
                                                 const float* __restrict__ qsum,
                                                 const unsigned short* __restrict__ kvt_u16,
                                                 float* __restrict__ out) {
    __shared__ __align__(16) unsigned char smB[64 * 1024];   // 128 e-rows x 256 i (bf16)

    int blk = blockIdx.x;
    const int nb = blk & 1;   blk >>= 1;
    const int mb = blk & 63;  blk >>= 6;
    const int bh = blk;
    const int cb = (bh >> 3) * CC + (bh & 7) * DD;

    const int wave = threadIdx.x >> 5, lane = threadIdx.x & 31;
    const int wm = wave & 3, wn = wave >> 2;
    const int eBase = nb * 128;                 // block's output-feature tile base
    const int n0 = mb * 128 + wm * 32;          // sequence origin of wave
    const int eW = wn * 64;                     // wave's e-offset inside LDS tile
    const int lr = lane & 15, hp = lane >> 4;

    // ---- stage kv^T[eBase .. eBase+128) x [0..256) into LDS (async DMA) ----
    unsigned short* gsrc = const_cast<unsigned short*>(kvt_u16) + ((size_t)bh * DD + eBase) * DD;
    for (int c = threadIdx.x; c < 4096; c += 256) {          // 4096 x 16B = 64KB
        char* ga = (char*)gsrc + (size_t)c * 16;
        unsigned char* la = smB + c * 16;
#if __has_builtin(__builtin_amdgcn_global_load_async_to_lds_b128)
        __builtin_amdgcn_global_load_async_to_lds_b128(
            (gptr128)ga, (lptr128)la, 0, 0);
#else
        *(float4*)la = *(const float4*)ga;
#endif
    }
#if __has_builtin(__builtin_amdgcn_s_wait_asynccnt)
    __builtin_amdgcn_s_wait_asynccnt(0);
#else
    asm volatile("s_wait_asynccnt 0" ::: "memory");
#endif
    __syncthreads();
    const __bf16* smb = (const __bf16*)smB;

    float mxA[2], rsA[2];
    const float* qcol[2];
#pragma unroll
    for (int mt = 0; mt < 2; ++mt) {
        const int nl = n0 + 16 * mt + lr;
        mxA[mt]  = qmax[(size_t)bh * NN + nl];
        rsA[mt]  = 1.0f / qsum[(size_t)bh * NN + nl];
        qcol[mt] = q + (size_t)cb * NN + nl;
    }

    v8f acc[2][4] = {};

    for (int i0 = 0; i0 < DD; i0 += 32) {
        v16bf aF[2];
#pragma unroll
        for (int mt = 0; mt < 2; ++mt) {
#pragma unroll
            for (int j = 0; j < 16; ++j) {
                const int i = i0 + ((j < 8) ? (8 * hp + j) : (8 + 8 * hp + j));
                const float x = qcol[mt][(size_t)i * NN];
                aF[mt][j] = f2bf(__expf(x - mxA[mt]) * rsA[mt]);
            }
        }
        v16bf bF[4];
#pragma unroll
        for (int nt = 0; nt < 4; ++nt) {
            const __bf16* rp = smb + ((eW + 16 * nt + lr) * DD + i0 + 8 * hp);
            v8bf lo = *(const v8bf*)(rp);
            v8bf hi = *(const v8bf*)(rp + 16);
#pragma unroll
            for (int j = 0; j < 8; ++j) { bF[nt][j] = lo[j]; bF[nt][8 + j] = hi[j]; }
        }
#pragma unroll
        for (int mt = 0; mt < 2; ++mt)
#pragma unroll
            for (int nt = 0; nt < 4; ++nt)
                acc[mt][nt] = __builtin_amdgcn_wmma_f32_16x16x32_bf16(
                    false, aF[mt], false, bF[nt], (short)0, acc[mt][nt], false, false);
    }

    // Store in final (B, H*d, N) layout: lane owns column e, 8 contiguous n's
#pragma unroll
    for (int mt = 0; mt < 2; ++mt)
#pragma unroll
        for (int nt = 0; nt < 4; ++nt) {
            const int e = eBase + eW + 16 * nt + lr;
            const int nbase = n0 + 16 * mt + 8 * hp;
            float* op = out + (size_t)(cb + e) * NN + nbase;
            *(float4*)(op)     = make_float4(acc[mt][nt][0], acc[mt][nt][1],
                                             acc[mt][nt][2], acc[mt][nt][3]);
            *(float4*)(op + 4) = make_float4(acc[mt][nt][4], acc[mt][nt][5],
                                             acc[mt][nt][6], acc[mt][nt][7]);
        }
}

// ---------------------------------------------------------------------------
extern "C" void kernel_launch(void* const* d_in, const int* in_sizes, int n_in,
                              void* d_out, int out_size, void* d_ws, size_t ws_size,
                              hipStream_t stream) {
    const float* q = (const float*)d_in[0];
    const float* k = (const float*)d_in[1];
    const float* v = (const float*)d_in[2];
    float* out = (float*)d_out;

    char* ws = (char*)d_ws;
    float* kmax = (float*)(ws);                                  //  32 KB
    float* ksum = (float*)(ws + 32768);                          //  32 KB
    float* qmax = (float*)(ws + 65536);                          //   1 MB
    float* qsum = (float*)(ws + 65536 + 1048576);                //   1 MB
    float* kv   = (float*)(ws + 2162688);                        //   8 MB
    unsigned short* kvt = (unsigned short*)(ws + 2162688 + 8388608); // 4 MB

    (void)hipMemsetAsync(kv, 0, (size_t)BH * DD * DD * sizeof(float), stream);

    k_rowstats  <<<BB * CC,              256, 0, stream>>>(k, kmax, ksum);
    q_colstats  <<<BH * (NN / 256),      256, 0, stream>>>(q, qmax, qsum);
    gemm1_kv    <<<BH * 2 * 2 * 8,       256, 0, stream>>>(k, v, kmax, ksum, kv);
    kv_transpose<<<(BH * DD * DD) / 256, 256, 0, stream>>>(kv, kvt);
    gemm2_out   <<<BH * 64 * 2,          256, 0, stream>>>(q, qmax, qsum, kvt, out);
}
